// SwinTransformerBlock_32521492365833
// MI455X (gfx1250) — matrix operand order repaired and checked by reading, not compiled, not run on hardware
//
#include <hip/hip_runtime.h>
#include <hip/hip_bf16.h>

// ---------------------------------------------------------------------------
// Swin block on gfx1250: all GEMM-shaped math via v_wmma_f32_16x16x32_f16.
// ---------------------------------------------------------------------------

typedef __attribute__((ext_vector_type(16))) _Float16 v16h;
typedef __attribute__((ext_vector_type(8)))  float    v8f;

#define DIM_   384
#define HID_   1536
#define HEADS_ 12
#define HW_    56
#define NWIN_  2048     // 32 images * 64 windows
#define NP_    64       // window rows padded 49->64
#define ROWS_  131072   // NWIN_*NP_
#define NTOK_  100352   // 32*56*56

__device__ __forceinline__ v8f wmma16(v16h a, v16h b, v8f c) {
  return __builtin_amdgcn_wmma_f32_16x16x32_f16(false, a, false, b, (short)0, c,
                                                false, false);
}

// A-fragment (16x32 f16): lane holds row (l&15); halves K in [8h,8h+8) and
// [16+8h, 16+8h+8)  -> two b128 loads from a row with contiguous K.
__device__ __forceinline__ v16h ldA(const _Float16* rowk, int h) {
  union { v16h v; uint4 u[2]; } U;
  U.u[0] = *(const uint4*)(rowk + 8 * h);
  U.u[1] = *(const uint4*)(rowk + 16 + 8 * h);
  return U.v;
}
// B-fragment (32x16 f16): lane holds col (l&15); halves K in [8h,8h+8) and
// [16+8h, 16+8h+8) (same order as A).
__device__ __forceinline__ v16h ldB(const _Float16* colk, int h) {
  union { v16h v; uint4 u[2]; } U;
  U.u[0] = *(const uint4*)(colk + 8 * h);
  U.u[1] = *(const uint4*)(colk + 16 + 8 * h);
  return U.v;
}

// ---------------------------------------------------------------------------
__global__ __launch_bounds__(256) void k_cast_f16(const float* __restrict__ s,
                                                  _Float16* __restrict__ d, int n) {
  int i = blockIdx.x * 256 + threadIdx.x;
  if (i < n) d[i] = (_Float16)s[i];
}

// LN1 + roll(-3,-3) + window partition -> f16, rows padded to 64 (zeros).
__global__ __launch_bounds__(256) void k_ln1_window(const float* __restrict__ x,
                                                    const float* __restrict__ w,
                                                    const float* __restrict__ b,
                                                    _Float16* __restrict__ Xw) {
  int lane = threadIdx.x & 31, wv = threadIdx.x >> 5;
  long idx = (long)blockIdx.x * 8 + wv;            // (window,row) 0..131071
  long win = idx >> 6; int n = (int)(idx & 63);
  _Float16* dst = Xw + idx * DIM_;
  if (n >= 49) {
#pragma unroll
    for (int j = 0; j < 12; ++j) dst[lane + 32 * j] = (_Float16)0.f;
    return;
  }
  int bb = (int)(win >> 6), wi = (int)(win & 63);
  int wh = wi >> 3, ww = wi & 7;
  int r = n / 7, cc = n - r * 7;
  int sh = (wh * 7 + r + 3) % HW_;                 // roll by -3
  int sw = (ww * 7 + cc + 3) % HW_;
  const float* src = x + ((long)(bb * HW_ + sh) * HW_ + sw) * DIM_;
  float vals[12], s = 0.f, s2 = 0.f;
#pragma unroll
  for (int j = 0; j < 12; ++j) {
    float v = src[lane + 32 * j];
    vals[j] = v; s += v; s2 += v * v;
  }
#pragma unroll
  for (int off = 16; off; off >>= 1) {
    s  += __shfl_xor(s,  off);
    s2 += __shfl_xor(s2, off);
  }
  float mean = s * (1.f / DIM_);
  float var  = s2 * (1.f / DIM_) - mean * mean;
  float rs   = rsqrtf(var + 1e-5f);
#pragma unroll
  for (int j = 0; j < 12; ++j) {
    int c = lane + 32 * j;
    dst[c] = (_Float16)((vals[j] - mean) * rs * w[c] + b[c]);
  }
}

// QKV GEMM: [131072x384] @ [384x1152]^T, scatter to Q,K (row) / V (transposed).
__global__ __launch_bounds__(256) void k_gemm_qkv(const _Float16* __restrict__ A,
                                                  const _Float16* __restrict__ W,
                                                  const float* __restrict__ bias,
                                                  _Float16* __restrict__ Qb,
                                                  _Float16* __restrict__ Kb,
                                                  _Float16* __restrict__ Vt) {
  int lane = threadIdx.x & 31, wv = threadIdx.x >> 5;
  int h = lane >> 4, nl = lane & 15;
  long mbase = (long)blockIdx.y * 128 + wv * 16;
  int obase = blockIdx.x * 64;
  const _Float16* arow = A + (mbase + nl) * DIM_;
  v8f acc[4] = {};
  for (int kb = 0; kb < DIM_; kb += 32) {
    __builtin_prefetch(arow + kb + 64, 0, 1);
    v16h a = ldA(arow + kb, h);
#pragma unroll
    for (int t = 0; t < 4; ++t) {
      v16h bfr = ldB(W + (long)(obase + 16 * t + nl) * DIM_ + kb, h);
      acc[t] = wmma16(a, bfr, acc[t]);
    }
  }
#pragma unroll
  for (int t = 0; t < 4; ++t) {
    int o = obase + 16 * t + nl;
    int which = o / DIM_, c = o - which * DIM_;
    int hd = c >> 5, d = c & 31;
    float bv = bias[o];
#pragma unroll
    for (int r = 0; r < 8; ++r) {
      long row = mbase + r + 8 * h;
      long win = row >> 6; int n = (int)(row & 63);
      float v = acc[t][r] + bv;
      if (which == 0) {
        v *= 0.1767766952966369f;  // 1/sqrt(32) folded into Q
        Qb[((win * HEADS_ + hd) * NP_ + n) * 32 + d] = (_Float16)v;
      } else if (which == 1) {
        Kb[((win * HEADS_ + hd) * NP_ + n) * 32 + d] = (_Float16)v;
      } else {
        Vt[((win * HEADS_ + hd) * 32 + d) * NP_ + n] = (_Float16)v;
      }
    }
  }
}

__device__ __forceinline__ int reg3(int p) { return p < 49 ? 0 : (p < 53 ? 1 : 2); }

// Attention: one block per (window, head). QK^T -> bias+mask -> softmax -> PV.
__global__ __launch_bounds__(128) void k_attn(const _Float16* __restrict__ Qb,
                                              const _Float16* __restrict__ Kb,
                                              const _Float16* __restrict__ Vt,
                                              const float* __restrict__ rpb,
                                              _Float16* __restrict__ Out) {
  __shared__ float S[64 * 65];
  __shared__ __align__(16) _Float16 P[64 * 80];
  int head = blockIdx.x, win = blockIdx.y;
  int tid = threadIdx.x, lane = tid & 31, wv = tid >> 5;
  int h = lane >> 4, nl = lane & 15;
  const _Float16* Qh = Qb + (long)(win * HEADS_ + head) * NP_ * 32;
  const _Float16* Kh = Kb + (long)(win * HEADS_ + head) * NP_ * 32;
  const _Float16* Vh = Vt + (long)(win * HEADS_ + head) * 32 * NP_;

  // S = Q K^T  (K=32 is one chunk); wave wv owns rows [16wv, 16wv+16)
  v8f acc[4] = {};
  {
    v16h a = ldA(Qh + (wv * 16 + nl) * 32, h);
#pragma unroll
    for (int t = 0; t < 4; ++t) {
      v16h bfr = ldB(Kh + (16 * t + nl) * 32, h);
      acc[t] = wmma16(a, bfr, acc[t]);
    }
  }
  int wi = win & 63, wh = wi >> 3, ww = wi & 7;
#pragma unroll
  for (int t = 0; t < 4; ++t) {
    int n = 16 * t + nl;
#pragma unroll
    for (int r = 0; r < 8; ++r) {
      int m = wv * 16 + r + 8 * h;
      float s;
      if (n < 49) {
        s = acc[t][r];
        if (m < 49) {
          int rm = m / 7, cm = m - rm * 7, rn = n / 7, cn = n - rn * 7;
          s += rpb[((rm - rn + 6) * 13 + (cm - cn + 6)) * HEADS_ + head];
          int cm_reg = reg3(wh * 7 + rm) * 3 + reg3(ww * 7 + cm);
          int cn_reg = reg3(wh * 7 + rn) * 3 + reg3(ww * 7 + cn);
          if (cm_reg != cn_reg) s -= 100.f;
        }
      } else {
        s = -1e30f;
      }
      S[m * 65 + n] = s;
    }
  }
  __syncthreads();
  if (tid < 64) {
    float mx = S[tid * 65];
    for (int c = 1; c < 64; ++c) mx = fmaxf(mx, S[tid * 65 + c]);
    float sum = 0.f;
    for (int c = 0; c < 64; ++c) {
      float e = __expf(S[tid * 65 + c] - mx);
      S[tid * 65 + c] = e; sum += e;
    }
    float inv = 1.f / sum;
    for (int c = 0; c < 64; ++c) P[tid * 80 + c] = (_Float16)(S[tid * 65 + c] * inv);
  }
  __syncthreads();

  // O = P @ V : M=64, N=32 (2 tiles), K=64 (2 chunks); A-frags from LDS.
  v8f o[2] = {};
#pragma unroll
  for (int kb = 0; kb < 64; kb += 32) {
    v16h a = ldA(P + (wv * 16 + nl) * 80 + kb, h);
#pragma unroll
    for (int t = 0; t < 2; ++t) {
      v16h bfr = ldB(Vh + (16 * t + nl) * NP_ + kb, h);
      o[t] = wmma16(a, bfr, o[t]);
    }
  }
#pragma unroll
  for (int t = 0; t < 2; ++t) {
    int d = 16 * t + nl;
#pragma unroll
    for (int r = 0; r < 8; ++r) {
      int m = wv * 16 + r + 8 * h;
      if (m < 49)
        Out[((long)win * NP_ + m) * DIM_ + head * 32 + d] = (_Float16)o[t][r];
    }
  }
}

// Proj GEMM + window-reverse + roll(+3,+3) + residual -> x1 (fp32).
__global__ __launch_bounds__(256) void k_gemm_proj(const _Float16* __restrict__ A,
                                                   const _Float16* __restrict__ W,
                                                   const float* __restrict__ bias,
                                                   const float* __restrict__ x,
                                                   float* __restrict__ x1) {
  int lane = threadIdx.x & 31, wv = threadIdx.x >> 5;
  int h = lane >> 4, nl = lane & 15;
  long mbase = (long)blockIdx.y * 128 + wv * 16;
  int obase = blockIdx.x * 64;
  const _Float16* arow = A + (mbase + nl) * DIM_;
  v8f acc[4] = {};
  for (int kb = 0; kb < DIM_; kb += 32) {
    __builtin_prefetch(arow + kb + 64, 0, 1);
    v16h a = ldA(arow + kb, h);
#pragma unroll
    for (int t = 0; t < 4; ++t) {
      v16h bfr = ldB(W + (long)(obase + 16 * t + nl) * DIM_ + kb, h);
      acc[t] = wmma16(a, bfr, acc[t]);
    }
  }
#pragma unroll
  for (int r = 0; r < 8; ++r) {
    long row = mbase + r + 8 * h;
    long win = row >> 6; int n = (int)(row & 63);
    if (n >= 49) continue;
    int bb = (int)(win >> 6), wi = (int)(win & 63);
    int wh = wi >> 3, ww = wi & 7;
    int rr = n / 7, cc = n - rr * 7;
    int dh = (wh * 7 + rr + 3) % HW_;              // roll back by +3
    int dw = (ww * 7 + cc + 3) % HW_;
    long tok = (long)(bb * HW_ + dh) * HW_ + dw;
#pragma unroll
    for (int t = 0; t < 4; ++t) {
      int c = obase + 16 * t + nl;
      x1[tok * DIM_ + c] = x[tok * DIM_ + c] + acc[t][r] + bias[c];
    }
  }
}

// LN2 over x1 -> f16.
__global__ __launch_bounds__(256) void k_ln2(const float* __restrict__ x1,
                                             const float* __restrict__ w,
                                             const float* __restrict__ b,
                                             _Float16* __restrict__ H16) {
  int lane = threadIdx.x & 31, wv = threadIdx.x >> 5;
  long idx = (long)blockIdx.x * 8 + wv;
  const float* src = x1 + idx * DIM_;
  _Float16* dst = H16 + idx * DIM_;
  float vals[12], s = 0.f, s2 = 0.f;
#pragma unroll
  for (int j = 0; j < 12; ++j) {
    float v = src[lane + 32 * j];
    vals[j] = v; s += v; s2 += v * v;
  }
#pragma unroll
  for (int off = 16; off; off >>= 1) {
    s  += __shfl_xor(s,  off);
    s2 += __shfl_xor(s2, off);
  }
  float mean = s * (1.f / DIM_);
  float var  = s2 * (1.f / DIM_) - mean * mean;
  float rs   = rsqrtf(var + 1e-5f);
#pragma unroll
  for (int j = 0; j < 12; ++j) {
    int c = lane + 32 * j;
    dst[c] = (_Float16)((vals[j] - mean) * rs * w[c] + b[c]);
  }
}

// FC1 GEMM + exact GELU -> f16.
__global__ __launch_bounds__(256) void k_gemm_fc1(const _Float16* __restrict__ A,
                                                  const _Float16* __restrict__ W,
                                                  const float* __restrict__ bias,
                                                  _Float16* __restrict__ O) {
  int lane = threadIdx.x & 31, wv = threadIdx.x >> 5;
  int h = lane >> 4, nl = lane & 15;
  long mbase = (long)blockIdx.y * 128 + wv * 16;
  int obase = blockIdx.x * 64;
  const _Float16* arow = A + (mbase + nl) * DIM_;
  v8f acc[4] = {};
  for (int kb = 0; kb < DIM_; kb += 32) {
    __builtin_prefetch(arow + kb + 64, 0, 1);
    v16h a = ldA(arow + kb, h);
#pragma unroll
    for (int t = 0; t < 4; ++t) {
      v16h bfr = ldB(W + (long)(obase + 16 * t + nl) * DIM_ + kb, h);
      acc[t] = wmma16(a, bfr, acc[t]);
    }
  }
#pragma unroll
  for (int t = 0; t < 4; ++t) {
    int o = obase + 16 * t + nl;
    float bv = bias[o];
#pragma unroll
    for (int r = 0; r < 8; ++r) {
      long row = mbase + r + 8 * h;
      float v = acc[t][r] + bv;
      v = 0.5f * v * (1.f + erff(v * 0.70710678118654752f));  // exact GELU
      O[row * HID_ + o] = (_Float16)v;
    }
  }
}

// FC2 GEMM + residual -> fp32 output.
__global__ __launch_bounds__(256) void k_gemm_fc2(const _Float16* __restrict__ A,
                                                  const _Float16* __restrict__ W,
                                                  const float* __restrict__ bias,
                                                  const float* __restrict__ x1,
                                                  float* __restrict__ out) {
  int lane = threadIdx.x & 31, wv = threadIdx.x >> 5;
  int h = lane >> 4, nl = lane & 15;
  long mbase = (long)blockIdx.y * 128 + wv * 16;
  int obase = blockIdx.x * 64;
  const _Float16* arow = A + (mbase + nl) * HID_;
  v8f acc[4] = {};
  for (int kb = 0; kb < HID_; kb += 32) {
    __builtin_prefetch(arow + kb + 64, 0, 1);
    v16h a = ldA(arow + kb, h);
#pragma unroll
    for (int t = 0; t < 4; ++t) {
      v16h bfr = ldB(W + (long)(obase + 16 * t + nl) * HID_ + kb, h);
      acc[t] = wmma16(a, bfr, acc[t]);
    }
  }
#pragma unroll
  for (int t = 0; t < 4; ++t) {
    int c = obase + 16 * t + nl;
    float bv = bias[c];
#pragma unroll
    for (int r = 0; r < 8; ++r) {
      long row = mbase + r + 8 * h;
      out[row * DIM_ + c] = x1[row * DIM_ + c] + acc[t][r] + bv;
    }
  }
}

// ---------------------------------------------------------------------------
extern "C" void kernel_launch(void* const* d_in, const int* in_sizes, int n_in,
                              void* d_out, int out_size, void* d_ws, size_t ws_size,
                              hipStream_t stream) {
  (void)in_sizes; (void)n_in; (void)out_size; (void)ws_size;
  const float* x      = (const float*)d_in[0];
  const float* n1w    = (const float*)d_in[1];
  const float* n1b    = (const float*)d_in[2];
  const float* qkv_w  = (const float*)d_in[3];
  const float* qkv_b  = (const float*)d_in[4];
  const float* rpb    = (const float*)d_in[5];
  const float* proj_w = (const float*)d_in[6];
  const float* proj_b = (const float*)d_in[7];
  const float* n2w    = (const float*)d_in[8];
  const float* n2b    = (const float*)d_in[9];
  const float* fc1_w  = (const float*)d_in[10];
  const float* fc1_b  = (const float*)d_in[11];
  const float* fc2_w  = (const float*)d_in[12];
  const float* fc2_b  = (const float*)d_in[13];

  char* ws = (char*)d_ws;
  // f16 weight copies (3.4 MB)
  const size_t o_wq = 0;
  const size_t o_wp = o_wq + (size_t)3 * DIM_ * DIM_ * 2;   //  884736
  const size_t o_w1 = o_wp + (size_t)DIM_ * DIM_ * 2;       // +294912
  const size_t o_w2 = o_w1 + (size_t)HID_ * DIM_ * 2;       // +1179648
  const size_t o_A  = o_w2 + (size_t)HID_ * DIM_ * 2;       // big region
  const size_t szQ  = (size_t)NWIN_ * HEADS_ * NP_ * 32 * 2;   // 100.66 MB each
  const size_t szA  = (size_t)NTOK_ * HID_ * 2;                // 308.28 MB (fc1out)
  const size_t o_Q  = o_A + szA;
  const size_t o_K  = o_Q + szQ;
  const size_t o_V  = o_K + szQ;

  _Float16* wq16 = (_Float16*)(ws + o_wq);
  _Float16* wp16 = (_Float16*)(ws + o_wp);
  _Float16* w116 = (_Float16*)(ws + o_w1);
  _Float16* w216 = (_Float16*)(ws + o_w2);
  _Float16* Xw   = (_Float16*)(ws + o_A);   // [2048][64][384] f16, then attnOut, then fc1out
  _Float16* Qb   = (_Float16*)(ws + o_Q);
  _Float16* Kb   = (_Float16*)(ws + o_K);
  _Float16* Vt   = (_Float16*)(ws + o_V);
  float*    x1   = (float*)   (ws + o_Q);   // reuse Q/K region after attention
  _Float16* H16  = (_Float16*)(ws + o_V);   // reuse V region after attention

  // 0) weight casts
  int nq = 3 * DIM_ * DIM_, np = DIM_ * DIM_, nh = HID_ * DIM_;
  k_cast_f16<<<(nq + 255) / 256, 256, 0, stream>>>(qkv_w,  wq16, nq);
  k_cast_f16<<<(np + 255) / 256, 256, 0, stream>>>(proj_w, wp16, np);
  k_cast_f16<<<(nh + 255) / 256, 256, 0, stream>>>(fc1_w,  w116, nh);
  k_cast_f16<<<(nh + 255) / 256, 256, 0, stream>>>(fc2_w,  w216, nh);

  // 1) LN1 + shift + window partition
  k_ln1_window<<<ROWS_ / 8, 256, 0, stream>>>(x, n1w, n1b, Xw);
  // 2) QKV GEMM
  k_gemm_qkv<<<dim3(18, ROWS_ / 128), 256, 0, stream>>>(Xw, wq16, qkv_b, Qb, Kb, Vt);
  // 3) windowed attention (reuses Xw region as output)
  k_attn<<<dim3(HEADS_, NWIN_), 128, 0, stream>>>(Qb, Kb, Vt, rpb, Xw);
  // 4) proj + reverse-shift + residual
  k_gemm_proj<<<dim3(6, ROWS_ / 128), 256, 0, stream>>>(Xw, wp16, proj_b, x, x1);
  // 5) LN2
  k_ln2<<<NTOK_ / 8, 256, 0, stream>>>(x1, n2w, n2b, H16);
  // 6) FC1 + GELU (output overwrites Xw region)
  k_gemm_fc1<<<dim3(24, NTOK_ / 128), 256, 0, stream>>>(H16, w116, fc1_b, Xw);
  // 7) FC2 + residual -> out
  k_gemm_fc2<<<dim3(6, NTOK_ / 128), 256, 0, stream>>>(Xw, w216, fc2_b, x1,
                                                       (float*)d_out);
}
